// BiMambaOperator_70033736729113
// MI455X (gfx1250) — matrix operand order repaired and checked by reading, not compiled, not run on hardware
//
#include <hip/hip_runtime.h>
#include <hip/hip_bf16.h>
#include <math.h>

// ---- problem constants (match reference) ----
static constexpr int SB   = 4;      // batch
static constexpr int SL   = 2048;   // seq len
static constexpr int SD   = 768;    // d_model
static constexpr int SDIN = 1536;   // d_inner
static constexpr int SN   = 16;     // d_state
static constexpr int SR   = 48;     // dt_rank
static constexpr int SBL  = SB * SL;        // 8192 rows
static constexpr int SXZ  = 2 * SDIN;       // 3072
static constexpr int SDBC = SR + 2 * SN;    // 80

typedef __bf16 bf16_t;
typedef __attribute__((ext_vector_type(16))) __bf16 v16bf;
typedef __attribute__((ext_vector_type(8)))  __bf16 v8bf;
typedef __attribute__((ext_vector_type(8)))  float  v8f;
typedef __attribute__((ext_vector_type(4)))  unsigned int u32x4;
typedef __attribute__((ext_vector_type(4)))  int i32x4;
typedef __attribute__((ext_vector_type(8)))  int i32x8;

// Tensor Data Mover: enabled; arity differs across toolchains.
#define USE_TDM 1
#if __has_include(<hip/amd_detail/amd_gfx1250_TDM.h>)
#define TDM_ARGS6 1
#else
#define TDM_ARGS6 0
#endif

// ---------------- fp32 -> bf16 convert (weights, once per launch) ----------
__global__ __launch_bounds__(256) void f32_to_bf16_kernel(
    const float* __restrict__ src, bf16_t* __restrict__ dst, int n)
{
    int i = blockIdx.x * 256 + threadIdx.x;
    const int stride = gridDim.x * 256;
    for (; i < n; i += stride) dst[i] = (bf16_t)src[i];
}

// ---------------- LayerNorm (bf16 output; only GEMMs consume xn) -----------
__global__ __launch_bounds__(256) void ln_kernel(
    const float* __restrict__ x, const float* __restrict__ g,
    const float* __restrict__ b, bf16_t* __restrict__ xn_bf)
{
    const int row = blockIdx.x;            // 0..SBL-1
    const int tid = threadIdx.x;
    const float* xr = x + (size_t)row * SD;
    __shared__ float red[256];

    float s = 0.f;
    for (int i = tid; i < SD; i += 256) s += xr[i];
    red[tid] = s; __syncthreads();
    for (int off = 128; off > 0; off >>= 1) {
        if (tid < off) red[tid] += red[tid + off];
        __syncthreads();
    }
    const float mu = red[0] * (1.f / SD);
    __syncthreads();

    float v = 0.f;
    for (int i = tid; i < SD; i += 256) { float d = xr[i] - mu; v += d * d; }
    red[tid] = v; __syncthreads();
    for (int off = 128; off > 0; off >>= 1) {
        if (tid < off) red[tid] += red[tid + off];
        __syncthreads();
    }
    const float rstd = rsqrtf(red[0] * (1.f / SD) + 1e-5f);

    bf16_t* xo = xn_bf + (size_t)row * SD;
    for (int i = tid; i < SD; i += 256)
        xo[i] = (bf16_t)((xr[i] - mu) * rstd * g[i] + b[i]);
}

// ---------------- WMMA GEMM: C[M,N] = Abf[M,K] * Wbf[N,K]^T (+epilogue) ----
// mode 0: plain; mode 1: softplus(acc + bias[n]); mode 2: acc + bias[n] + resid[m,n]
// Stores fp32 to C (if non-null) and bf16 to Cb (if non-null), both with ldc.
static constexpr int BM = 128, BN = 128, BK = 32;

#if USE_TDM
// One 2-D TDM descriptor: copy bf16 tile [<=128 rows][32 cols] (row stride
// ld elems) into LDS at lds_off; OOB rows/cols are zero-filled by the TDM.
__device__ __forceinline__ void tdm_load_tile(
    const bf16_t* gsrc, unsigned lds_off, int rem_k, int rem_rows, int ld)
{
    unsigned long long ga = (unsigned long long)(uintptr_t)gsrc;
    u32x4 g0;
    g0[0] = 1u;                                         // count=1, user mode
    g0[1] = lds_off;                                    // LDS byte address
    g0[2] = (unsigned)(ga & 0xFFFFFFFFull);             // global_addr[31:0]
    g0[3] = (unsigned)((ga >> 32) & 0x01FFFFFFull)      // global_addr[56:32]
          | (2u << 30);                                 // type = 2 (image)
    i32x8 g1;
    g1[0] = (int)(1u << 16);                            // data_size=1 -> 2 bytes
    g1[1] = (int)(((unsigned)rem_k & 0xFFFFu) << 16);   // tensor_dim0[15:0]
    g1[2] = (int)((((unsigned)rem_k >> 16) & 0xFFFFu)   // tensor_dim0[31:16]
          | (((unsigned)rem_rows & 0xFFFFu) << 16));    // tensor_dim1[15:0]
    g1[3] = (int)((((unsigned)rem_rows >> 16) & 0xFFFFu)// tensor_dim1[31:16]
          | ((unsigned)BK << 16));                      // tile_dim0 = 32
    g1[4] = (int)(unsigned)BN;                          // tile_dim1 = 128
    g1[5] = (int)(unsigned)ld;                          // tensor_dim0_stride
    g1[6] = 0;
    g1[7] = 0;
    i32x4 z4 = {0, 0, 0, 0};
#if TDM_ARGS6
    i32x8 z8 = {0, 0, 0, 0, 0, 0, 0, 0};
    __builtin_amdgcn_tensor_load_to_lds(g0, g1, z4, z4, z8, 0);
#else
    __builtin_amdgcn_tensor_load_to_lds(g0, g1, z4, z4, 0);
#endif
}
#endif

__global__ __launch_bounds__(256) void wmma_gemm_kernel(
    const bf16_t* __restrict__ A, int lda,
    const bf16_t* __restrict__ W, int ldw,
    float* __restrict__ C, bf16_t* __restrict__ Cb, int ldc,
    int M, int N, int K,
    const float* __restrict__ bias,
    const float* __restrict__ resid, int ldr,
    int mode)
{
    __shared__ __align__(16) bf16_t As[2][BM][BK];   // double-buffered (2x8 KB)
    __shared__ __align__(16) bf16_t Bs[2][BN][BK];   // double-buffered (2x8 KB)

    const int tid  = threadIdx.x;
    const int wave = tid >> 5;
    const int lane = tid & 31;
    const int hi   = lane >> 4;
    const int lo16 = lane & 15;
    const int m0   = blockIdx.x * BM;
    const int n0   = blockIdx.y * BN;

    v8f acc[8] = {};

#if USE_TDM
    const unsigned as_off = (unsigned)(uintptr_t)(&As[0][0][0]);
    const unsigned bs_off = (unsigned)(uintptr_t)(&Bs[0][0][0]);
    const unsigned bufstep = (unsigned)(BM * BK * sizeof(bf16_t));   // 8 KB

    // prologue: fill buffer 0
    if (wave == 0) tdm_load_tile(W + (size_t)n0 * ldw, bs_off, K, N - n0, ldw);
    if (wave == 1) tdm_load_tile(A + (size_t)m0 * lda, as_off, K, M - m0, lda);
#endif

    int cur = 0;
    for (int kk = 0; kk < K; kk += BK) {
#if USE_TDM
        __builtin_amdgcn_s_wait_tensorcnt(0);   // issuing waves only; others free
        __syncthreads();                        // buffer 'cur' ready, prev reads done
        if (kk + BK < K) {                      // prefetch next tile into other buffer
            const int nxt = cur ^ 1;
            if (wave == 0)
                tdm_load_tile(W + (size_t)n0 * ldw + kk + BK,
                              bs_off + (unsigned)nxt * bufstep,
                              K - kk - BK, N - n0, ldw);
            if (wave == 1)
                tdm_load_tile(A + (size_t)m0 * lda + kk + BK,
                              as_off + (unsigned)nxt * bufstep,
                              K - kk - BK, M - m0, lda);
        }
#else
        __syncthreads();
        for (int i = tid; i < BN * BK; i += 256) {
            int n = i >> 5, k = i & 31;
            int gn = n0 + n, gk = kk + k;
            Bs[cur][n][k] = (gn < N && gk < K) ? W[(size_t)gn * ldw + gk] : (bf16_t)0.f;
        }
        for (int i = tid; i < BM * BK; i += 256) {
            int r = i >> 5, c = i & 31;
            int gm = m0 + r, gk = kk + c;
            As[cur][r][c] = (gm < M && gk < K) ? A[(size_t)gm * lda + gk] : (bf16_t)0.f;
        }
        __syncthreads();
#endif
        // A fragment (16x32, ISA 7.12.2): lane -> M=lane&15;
        // halves 0-7: k = hi*8.. ; halves 8-15: k = 16+hi*8.. -> 2x ds_load_b128
        const int mrow = (wave << 4) + lo16;
        const v8bf a_lo = *(const v8bf*)&As[cur][mrow][hi << 3];
        const v8bf a_hi = *(const v8bf*)&As[cur][mrow][16 + (hi << 3)];
        const v16bf af = __builtin_shufflevector(a_lo, a_hi,
            0, 1, 2, 3, 4, 5, 6, 7, 8, 9, 10, 11, 12, 13, 14, 15);

        // B fragments: lane -> N=lane&15; halves 0-7: k = hi*8.. ; halves 8-15: k = 16+hi*8.. -> 2x ds_load_b128
        #pragma unroll
        for (int j = 0; j < 8; ++j) {
            const int nrow = (j << 4) + lo16;
            const v8bf b_lo = *(const v8bf*)&Bs[cur][nrow][hi << 3];
            const v8bf b_hi = *(const v8bf*)&Bs[cur][nrow][16 + (hi << 3)];
            const v16bf bfr = __builtin_shufflevector(b_lo, b_hi,
                0, 1, 2, 3, 4, 5, 6, 7, 8, 9, 10, 11, 12, 13, 14, 15);
            acc[j] = __builtin_amdgcn_wmma_f32_16x16x32_bf16(
                false, af, false, bfr, (short)0, acc[j], false, false);
        }
        cur ^= 1;
    }

    // store: VGPR r -> (M = r + 8*hi, N = lane&15)
    const int rbase = m0 + (wave << 4) + (hi << 3);
    #pragma unroll
    for (int j = 0; j < 8; ++j) {
        const int col = n0 + (j << 4) + lo16;
        if (col >= N) continue;
        #pragma unroll
        for (int r = 0; r < 8; ++r) {
            const int row = rbase + r;
            if (row >= M) continue;
            float v = acc[j][r];
            if (mode == 1) {
                v += bias[col];
                v = (v > 20.f) ? v : log1pf(__expf(v));     // softplus
            } else if (mode == 2) {
                v += bias[col] + resid[(size_t)row * ldr + col];
            }
            if (C)  C[(size_t)row * ldc + col] = v;
            if (Cb) Cb[(size_t)row * ldc + col] = (bf16_t)v;
        }
    }
}

// ------- causal / anti-causal depthwise conv + SiLU (f32 + bf16 mirror) ----
__global__ __launch_bounds__(256) void conv_silu_kernel(
    const float* __restrict__ xz, const float* __restrict__ cw,
    const float* __restrict__ cb, float* __restrict__ u,
    bf16_t* __restrict__ u_bf, int dir)
{
    int idx = blockIdx.x * 256 + threadIdx.x;
    if (idx >= SBL * SDIN) return;
    const int c = idx % SDIN;
    const int t = (idx / SDIN) % SL;
    const int b = idx / (SDIN * SL);
    const float* w = cw + c * 4;

    float acc = cb[c];
    if (dir == 0) {                      // causal: xc[t-3+k] * w[k]
        #pragma unroll
        for (int k = 0; k < 4; ++k) {
            int tt = t - 3 + k;
            if (tt >= 0) acc += w[k] * xz[((size_t)(b * SL + tt)) * SXZ + c];
        }
    } else {                             // anti-causal (flip cancelled): xc[t+j]*w[3-j]
        #pragma unroll
        for (int j = 0; j < 4; ++j) {
            int tt = t + j;
            if (tt < SL) acc += w[3 - j] * xz[((size_t)(b * SL + tt)) * SXZ + c];
        }
    }
    const float s = acc / (1.f + __expf(-acc));   // SiLU
    const size_t o = (size_t)(b * SL + t) * SDIN + c;
    u[o] = s;
    u_bf[o] = (bf16_t)s;
}

// ------- selective scan + D skip + SiLU(z) gate (bf16 out for out_proj) ----
__global__ __launch_bounds__(256) void scan_kernel(
    const float* __restrict__ u, const float* __restrict__ delta,
    const float* __restrict__ dbc, const float* __restrict__ A_log,
    const float* __restrict__ Dp, const float* __restrict__ xz,
    bf16_t* __restrict__ y_bf, int dir)
{
    const int blk = blockIdx.x;                    // SB * (SDIN/256) = 24
    const int b = blk / (SDIN / 256);
    const int d = (blk % (SDIN / 256)) * 256 + threadIdx.x;

    float Ac[SN];
    #pragma unroll
    for (int n = 0; n < SN; ++n) Ac[n] = -__expf(A_log[(size_t)d * SN + n]);
    const float Dv = Dp[d];

    float h[SN];
    #pragma unroll
    for (int n = 0; n < SN; ++n) h[n] = 0.f;

    __shared__ float BC[2 * SN];                   // B: [0..15], C: [16..31]

    for (int s = 0; s < SL; ++s) {
        const int t = dir ? (SL - 1 - s) : s;
        __syncthreads();
        if (threadIdx.x < 2 * SN)
            BC[threadIdx.x] = dbc[(size_t)(b * SL + t) * SDBC + SR + threadIdx.x];
        __syncthreads();

        const size_t rowd = (size_t)(b * SL + t) * SDIN + d;
        const float dl = delta[rowd];
        const float uv = u[rowd];
        const float du = dl * uv;
        float ysum = 0.f;
        #pragma unroll
        for (int n = 0; n < SN; ++n) {
            const float dA = __expf(dl * Ac[n]);
            h[n] = dA * h[n] + du * BC[n];
            ysum += h[n] * BC[SN + n];
        }
        const float z = xz[(size_t)(b * SL + t) * SXZ + SDIN + d];
        const float gate = z / (1.f + __expf(-z));
        y_bf[rowd] = (bf16_t)((ysum + uv * Dv) * gate);
    }
}

// ---------------- launcher ----------------
extern "C" void kernel_launch(void* const* d_in, const int* in_sizes, int n_in,
                              void* d_out, int out_size, void* d_ws, size_t ws_size,
                              hipStream_t stream)
{
    const float* x         = (const float*)d_in[0];
    const float* in_proj_w = (const float*)d_in[1];   // (2, 3072, 768)
    const float* conv_w    = (const float*)d_in[2];   // (2, 1536, 4)
    const float* conv_b    = (const float*)d_in[3];   // (2, 1536)
    const float* x_proj_w  = (const float*)d_in[4];   // (2, 80, 1536)
    const float* dt_proj_w = (const float*)d_in[5];   // (2, 1536, 48)
    const float* dt_proj_b = (const float*)d_in[6];   // (2, 1536)
    const float* A_log     = (const float*)d_in[7];   // (2, 1536, 16)
    const float* D_param   = (const float*)d_in[8];   // (2, 1536)
    const float* out_proj_w= (const float*)d_in[9];   // (2, 768, 1536)
    const float* ln_g      = (const float*)d_in[10];  // (768,)
    const float* ln_b      = (const float*)d_in[11];  // (768,)
    const float* fusion_w  = (const float*)d_in[12];  // (768, 1536)
    const float* fusion_b  = (const float*)d_in[13];  // (768,)
    float* out = (float*)d_out;                       // (4, 2048, 768) fp32

    // workspace carve-up: fp32 region first, then bf16 region
    float* ws     = (float*)d_ws;
    float* xz     = ws;                                        // SBL*3072
    float* u      = xz    + (size_t)SBL * SXZ;                 // SBL*1536
    float* dbc    = u     + (size_t)SBL * SDIN;                // SBL*80
    float* delta  = dbc   + (size_t)SBL * SDBC;                // SBL*1536
    bf16_t* xn_bf    = (bf16_t*)(delta + (size_t)SBL * SDIN);  // SBL*768
    bf16_t* u_bf     = xn_bf    + (size_t)SBL * SD;            // SBL*1536
    bf16_t* dbc_bf   = u_bf     + (size_t)SBL * SDIN;          // SBL*80
    bf16_t* yb_bf    = dbc_bf   + (size_t)SBL * SDBC;          // SBL*1536
    bf16_t* outcat_bf= yb_bf    + (size_t)SBL * SDIN;          // SBL*1536
    bf16_t* w_in  = outcat_bf + (size_t)SBL * SDIN;            // 2*3072*768
    bf16_t* w_x   = w_in  + (size_t)2 * SXZ  * SD;             // 2*80*1536
    bf16_t* w_dt  = w_x   + (size_t)2 * SDBC * SDIN;           // 2*1536*48
    bf16_t* w_out = w_dt  + (size_t)2 * SDIN * SR;             // 2*768*1536
    bf16_t* w_fus = w_out + (size_t)2 * SD   * SDIN;           // 768*1536

    // pre-convert all weights to bf16 (TDM copies them verbatim into LDS)
    f32_to_bf16_kernel<<<512, 256, 0, stream>>>(in_proj_w,  w_in,  2 * SXZ * SD);
    f32_to_bf16_kernel<<<512, 256, 0, stream>>>(x_proj_w,   w_x,   2 * SDBC * SDIN);
    f32_to_bf16_kernel<<<512, 256, 0, stream>>>(dt_proj_w,  w_dt,  2 * SDIN * SR);
    f32_to_bf16_kernel<<<512, 256, 0, stream>>>(out_proj_w, w_out, 2 * SD * SDIN);
    f32_to_bf16_kernel<<<512, 256, 0, stream>>>(fusion_w,   w_fus, SD * SDIN);

    ln_kernel<<<SBL, 256, 0, stream>>>(x, ln_g, ln_b, xn_bf);

    for (int dir = 0; dir < 2; ++dir) {
        // in_proj: xz(8192x3072,f32) = xn_bf(8192x768) * W^T
        {
            dim3 g(SBL / BM, (SXZ + BN - 1) / BN);
            wmma_gemm_kernel<<<g, 256, 0, stream>>>(
                xn_bf, SD, w_in + (size_t)dir * SXZ * SD, SD,
                xz, nullptr, SXZ, SBL, SXZ, SD, nullptr, nullptr, 0, 0);
        }
        // depthwise conv + SiLU -> u (f32) + u_bf
        {
            int tot = SBL * SDIN;
            conv_silu_kernel<<<(tot + 255) / 256, 256, 0, stream>>>(
                xz, conv_w + (size_t)dir * SDIN * 4, conv_b + (size_t)dir * SDIN,
                u, u_bf, dir);
        }
        // x_proj: dbc(8192x80) = u_bf * W^T   (f32 for scan, bf16 for dt_proj)
        {
            dim3 g(SBL / BM, (SDBC + BN - 1) / BN);
            wmma_gemm_kernel<<<g, 256, 0, stream>>>(
                u_bf, SDIN, w_x + (size_t)dir * SDBC * SDIN, SDIN,
                dbc, dbc_bf, SDBC, SBL, SDBC, SDIN, nullptr, nullptr, 0, 0);
        }
        // dt_proj: delta(f32) = softplus(dbc_bf[:,0:48] * W^T + b)
        {
            dim3 g(SBL / BM, (SDIN + BN - 1) / BN);
            wmma_gemm_kernel<<<g, 256, 0, stream>>>(
                dbc_bf, SDBC, w_dt + (size_t)dir * SDIN * SR, SR,
                delta, nullptr, SDIN, SBL, SDIN, SR,
                dt_proj_b + (size_t)dir * SDIN, nullptr, 0, 1);
        }
        // selective scan + D skip + gate -> yb_bf
        scan_kernel<<<SB * (SDIN / 256), 256, 0, stream>>>(
            u, delta, dbc, A_log + (size_t)dir * SDIN * SN,
            D_param + (size_t)dir * SDIN, xz, yb_bf, dir);
        // out_proj into outcat_bf columns [dir*768, dir*768+768), ldc = 1536
        {
            dim3 g(SBL / BM, (SD + BN - 1) / BN);
            wmma_gemm_kernel<<<g, 256, 0, stream>>>(
                yb_bf, SDIN, w_out + (size_t)dir * SD * SDIN, SDIN,
                nullptr, outcat_bf + dir * SD, SDIN, SBL, SD, SDIN,
                nullptr, nullptr, 0, 0);
        }
    }

    // fusion: out(f32) = outcat_bf(8192x1536) * fusion_w^T + fusion_b + x
    {
        dim3 g(SBL / BM, (SD + BN - 1) / BN);
        wmma_gemm_kernel<<<g, 256, 0, stream>>>(
            outcat_bf, SDIN, w_fus, SDIN,
            out, nullptr, SD, SBL, SD, SDIN, fusion_b, x, SD, 2);
    }
}
